// MOLELayer_2826088481473
// MI455X (gfx1250) — compile-verified, matrix operand order checked by reading, not run on hardware
//
#include <hip/hip_runtime.h>
#include <hip/hip_bf16.h>
#include <math.h>

// ---------------------------------------------------------------------------
// MoLE layer for MI455X (gfx1250, wave32, WMMA).
//   out   = gelu(x @ base_w^T + base_b) + w_top1 * (gelu(x @ A_e) @ B_e)
//   probs = softmax(x @ gate_w^T + gate_b)
//
// Roofline: base GEMM 68.7 GF vs ~150 MB HBM traffic (~6.4 us @ 23.3 TB/s)
// -> compute bound -> bf16 WMMA (16x16x32) with fp32 accumulation.
//
// Expert trick: top-1 masked LoRA sum == dense rank-128 GEMM:
//   coef[t][e*16+r] = (e==idx[t]) ? w[t]*gelu(h[t][r]) : 0      ([T,128])
//   expert_out      = coef @ lora_B_all                          ([128,2048])
// lora_B ([E,R,D] row-major) IS the [128,2048] B matrix. gelu the
// accumulators in-register after the base K-steps, then 4 more WMMA K-steps
// into the SAME accumulators.
//
// Main loop is software-pipelined with double-buffered LDS: one barrier per
// K-step; next tile's global_load_b64s issue before the 8 WMMAs, and the
// fp32->bf16 convert + ds_store lands in the other buffer afterwards, so
// global latency hides behind matrix work.
// ---------------------------------------------------------------------------

#define DIM   2048
#define RANK  16
#define NEXP  8
#define T_TOK 8192
#define KLORA (NEXP * RANK)   // 128

typedef __attribute__((ext_vector_type(16))) __bf16 v16bf;
typedef __attribute__((ext_vector_type(8)))  float  v8f;

__device__ __forceinline__ unsigned int f32_to_bf16_rne(float f) {
    unsigned int u = __float_as_uint(f);
    return (u + 0x7FFFu + ((u >> 16) & 1u)) >> 16;
}
__device__ __forceinline__ unsigned int pack2bf(float lo, float hi) {
    return f32_to_bf16_rne(lo) | (f32_to_bf16_rne(hi) << 16);
}
__device__ __forceinline__ float gelu_exact(float v) {
    return 0.5f * v * (1.0f + erff(v * 0.70710678118654752440f));
}

// ---------------------------------------------------------------------------
// Kernel 1: router + selected-expert LoRA-A. One wave32 per token.
// ---------------------------------------------------------------------------
__global__ __launch_bounds__(256) void mole_router_kernel(
    const float* __restrict__ x, const float* __restrict__ gate_w,
    const float* __restrict__ gate_b, const float* __restrict__ lora_A,
    float* __restrict__ probs_out, float* __restrict__ coef_out)
{
    const int wid  = threadIdx.x >> 5;
    const int lane = threadIdx.x & 31;
    const int t    = blockIdx.x * 8 + wid;
    const float* xrow = x + (size_t)t * DIM;

    // gate logits: lane-strided dots, shfl_xor tree reduction
    float ge[NEXP];
#pragma unroll
    for (int e = 0; e < NEXP; ++e) ge[e] = 0.f;
    for (int k = lane; k < DIM; k += 32) {
        float xv = xrow[k];
#pragma unroll
        for (int e = 0; e < NEXP; ++e) ge[e] += xv * gate_w[e * DIM + k];
    }
#pragma unroll
    for (int e = 0; e < NEXP; ++e) {
#pragma unroll
        for (int off = 16; off > 0; off >>= 1)
            ge[e] += __shfl_xor(ge[e], off, 32);
        ge[e] += gate_b[e];
    }

    // softmax + first-occurrence argmax (matches jnp.argmax)
    float mx = ge[0]; int best = 0;
#pragma unroll
    for (int e = 1; e < NEXP; ++e) if (ge[e] > mx) { mx = ge[e]; best = e; }
    float pr[NEXP]; float se = 0.f;
#pragma unroll
    for (int e = 0; e < NEXP; ++e) { pr[e] = expf(ge[e] - mx); se += pr[e]; }
    const float inv = 1.f / se;
    if (lane < NEXP) probs_out[(size_t)t * NEXP + lane] = pr[lane] * inv;
    const float w = pr[best] * inv;

    // selected expert: h[r] = gelu(x . A_e[:,r]); lane r = lane&15, k parity split
    const int r    = lane & 15;
    const int half = lane >> 4;
    const float* Ae = lora_A + (size_t)best * DIM * RANK;
    float acc = 0.f;
    for (int k = half; k < DIM; k += 2)
        acc += xrow[k] * Ae[(size_t)k * RANK + r];
    acc += __shfl_xor(acc, 16, 32);            // all lanes: full sum for r=lane&15
    const float hval = gelu_exact(acc) * w;

    // dense coef row: 128 entries, zero except [best*16 .. best*16+15]
#pragma unroll
    for (int j = 0; j < 4; ++j) {
        const int k = lane + 32 * j;
        const float src = __shfl(hval, k & 15, 32);   // value for rank k&15
        coef_out[(size_t)t * KLORA + k] = ((k >> 4) == best) ? src : 0.f;
    }
}

// ---------------------------------------------------------------------------
// Kernel 2: fused GEMM. Block 128x128x32; 8 waves, each 2x4 tiles of 16x16.
// CDNA5 16-bit fragment layout: lane m/n = lane&15, half = lane>>4;
// reg v -> kpair kp = (v>>2)*8 + half*4 + (v&3). LDS strides padded (17/133).
// ---------------------------------------------------------------------------
#define BM 128
#define BN 128
#define BK 32
#define SA_STRIDE 17    // dwords per A row (16 k-pairs + pad)
#define SB_STRIDE 133   // dwords per B k-pair row (128 n + pad)

__global__ __launch_bounds__(256) void mole_gemm_kernel(
    const float* __restrict__ x, const float* __restrict__ base_w,
    const float* __restrict__ base_b, const float* __restrict__ lora_B,
    const float* __restrict__ coef, float* __restrict__ out)
{
    __shared__ unsigned int sA[2][BM * SA_STRIDE];        // 2 x 8704 B
    __shared__ unsigned int sB[2][(BK / 2) * SB_STRIDE];  // 2 x 8512 B

    const int tid  = threadIdx.x;
    const int lane = tid & 31;
    const int wid  = tid >> 5;
    const int lnid = lane & 15;
    const int half = lane >> 4;

    const int m0 = blockIdx.x * BM;   // token tile base
    const int n0 = blockIdx.y * BN;   // output-feature tile base

    const int waveM = wid >> 1;  // 0..3 -> 32 rows each
    const int waveN = wid & 1;   // 0..1 -> 64 cols each

    union Frag { unsigned int u[8]; v16bf v; };
    union Acc  { float f[8]; v8f v; };
    Acc acc[2][4];
#pragma unroll
    for (int i = 0; i < 2; ++i)
#pragma unroll
        for (int j = 0; j < 4; ++j)
#pragma unroll
            for (int r = 0; r < 8; ++r) acc[i][j].f[r] = 0.f;

    const int kpL = tid & 15;   // k-pair index for tile loading
    const int rL  = tid >> 4;   // row 0..15, strided by 16
    const int nB  = tid & 127;  // n index for lora_B loading
    const int kpB = tid >> 7;   // 0..1

    float2 ldA[8], ldB[8];

    // issue next tile's global loads into registers (no LDS yet)
    auto issue_base = [&](int kt) {
#pragma unroll
        for (int it = 0; it < 8; ++it) {     // A: x[m0+m][kt+2kp..+1]
            const int m = rL + it * 16;
            ldA[it] = *(const float2*)(x + (size_t)(m0 + m) * DIM + kt + 2 * kpL);
        }
#pragma unroll
        for (int it = 0; it < 8; ++it) {     // B: base_w[n0+n][kt+2kp..+1] (transpose)
            const int n = rL + it * 16;
            ldB[it] = *(const float2*)(base_w + (size_t)(n0 + n) * DIM + kt + 2 * kpL);
        }
    };
    // convert fp32->bf16 pairs and commit to LDS buffer `buf`
    auto commit_base = [&](int buf) {
#pragma unroll
        for (int it = 0; it < 8; ++it)
            sA[buf][(rL + it * 16) * SA_STRIDE + kpL] = pack2bf(ldA[it].x, ldA[it].y);
#pragma unroll
        for (int it = 0; it < 8; ++it)
            sB[buf][kpL * SB_STRIDE + (rL + it * 16)] = pack2bf(ldB[it].x, ldB[it].y);
    };
    // build fragments from buffer `buf` and run the 8 WMMAs
    auto compute = [&](int buf) {
        Frag aF[2], bF[4];
#pragma unroll
        for (int tm = 0; tm < 2; ++tm) {
            const int m = waveM * 32 + tm * 16 + lnid;
#pragma unroll
            for (int v = 0; v < 8; ++v) {
                const int kp = (v >> 2) * 8 + half * 4 + (v & 3);
                aF[tm].u[v] = sA[buf][m * SA_STRIDE + kp];
            }
        }
#pragma unroll
        for (int tn = 0; tn < 4; ++tn) {
            const int n = waveN * 64 + tn * 16 + lnid;
#pragma unroll
            for (int v = 0; v < 8; ++v) {
                const int kp = (v >> 2) * 8 + half * 4 + (v & 3);
                bF[tn].u[v] = sB[buf][kp * SB_STRIDE + n];
            }
        }
#pragma unroll
        for (int tm = 0; tm < 2; ++tm)
#pragma unroll
            for (int tn = 0; tn < 4; ++tn)
                acc[tm][tn].v = __builtin_amdgcn_wmma_f32_16x16x32_bf16(
                    false, aF[tm].v, false, bF[tn].v,
                    (short)0, acc[tm][tn].v, false, false);
    };

    // ---- Phase 1: base GEMM, K = 2048, double-buffered pipeline ----
    issue_base(0);
    commit_base(0);
    int cur = 0;
    for (int kt = 0; kt < DIM; kt += BK) {
        __syncthreads();                      // buf[cur] stores visible to all
        const bool more = (kt + BK) < DIM;
        if (more) issue_base(kt + BK);        // overlap with WMMAs below
        compute(cur);
        if (more) commit_base(cur ^ 1);       // waits loads, cvt, ds_store
        cur ^= 1;
    }

    // ---- gelu(base + bias) in-register ----
#pragma unroll
    for (int tn = 0; tn < 4; ++tn) {
        const float bb = base_b[n0 + waveN * 64 + tn * 16 + lnid];
#pragma unroll
        for (int tm = 0; tm < 2; ++tm)
#pragma unroll
            for (int r = 0; r < 8; ++r)
                acc[tm][tn].f[r] = gelu_exact(acc[tm][tn].f[r] + bb);
    }

    // ---- Phase 2: expert GEMM, K = 128, into the gelu'd accumulators ----
    for (int kb = 0; kb < KLORA; kb += BK) {
        __syncthreads();
#pragma unroll
        for (int it = 0; it < 8; ++it) {   // A: coef[m0+m][kb+2kp..+1]
            const int m = rL + it * 16;
            const float2 cv = *(const float2*)(coef + (size_t)(m0 + m) * KLORA + kb + 2 * kpL);
            sA[0][m * SA_STRIDE + kpL] = pack2bf(cv.x, cv.y);
        }
#pragma unroll
        for (int it = 0; it < 8; ++it) {   // B: lora_B is already [128][2048] row-major
            const int kp = kpB + 2 * it;   // 0..15
            const int k0 = kb + 2 * kp;
            const float lo = lora_B[(size_t)k0 * DIM + n0 + nB];
            const float hi = lora_B[(size_t)(k0 + 1) * DIM + n0 + nB];
            sB[0][kp * SB_STRIDE + nB] = pack2bf(lo, hi);
        }
        __syncthreads();
        compute(0);
    }

    // ---- store: C layout row m = r + 8*half, col n = lane&15 ----
#pragma unroll
    for (int tm = 0; tm < 2; ++tm) {
#pragma unroll
        for (int tn = 0; tn < 4; ++tn) {
            const int ng = n0 + waveN * 64 + tn * 16 + lnid;
#pragma unroll
            for (int r = 0; r < 8; ++r) {
                const int tg = m0 + waveM * 32 + tm * 16 + r + 8 * half;
                out[(size_t)tg * DIM + ng] = acc[tm][tn].f[r];
            }
        }
    }
}

// ---------------------------------------------------------------------------
// inputs (setup_inputs order): x, gate_w, gate_b, base_w, base_b, lora_A, lora_B
// d_out: out [8192*2048] fp32, then probs [8192*8] fp32
// d_ws: coef [8192*128] fp32 (4 MB)
// ---------------------------------------------------------------------------
extern "C" void kernel_launch(void* const* d_in, const int* in_sizes, int n_in,
                              void* d_out, int out_size, void* d_ws, size_t ws_size,
                              hipStream_t stream) {
    (void)in_sizes; (void)n_in; (void)out_size; (void)ws_size;
    const float* x      = (const float*)d_in[0];
    const float* gate_w = (const float*)d_in[1];
    const float* gate_b = (const float*)d_in[2];
    const float* base_w = (const float*)d_in[3];
    const float* base_b = (const float*)d_in[4];
    const float* lora_A = (const float*)d_in[5];
    const float* lora_B = (const float*)d_in[6];

    float* out   = (float*)d_out;
    float* probs = out + (size_t)T_TOK * DIM;
    float* coef  = (float*)d_ws;

    mole_router_kernel<<<T_TOK / 8, 256, 0, stream>>>(
        x, gate_w, gate_b, lora_A, probs, coef);

    dim3 grid(T_TOK / BM, DIM / BN);
    mole_gemm_kernel<<<grid, 256, 0, stream>>>(
        x, base_w, base_b, lora_B, coef, out);
}
